// CrossAttention_39032662786362
// MI455X (gfx1250) — matrix operand order repaired and checked by reading, not compiled, not run on hardware
//
#include <hip/hip_runtime.h>

// ---------------------------------------------------------------------------
// Windowed self-attention (reference "CrossAttention" with context-overwrite
// bug) for MI455X / gfx1250.  All matmuls run on v_wmma_f32_16x16x32_bf16
// (fp32 accumulate).  Pipeline:
//   1) Q/K/V projections: bf16 WMMA GEMM  (16384x1024 @ 1024x1024) x3,
//      double-buffered LDS tiles so global loads hide under WMMA issue
//   2) per-(window,head) attention, fully LDS-resident (224KB of 320KB/WGP),
//      Q/K staged with global_load_async_to_lds_b128 (ASYNCcnt path)
//   3) output projection + bias: bf16 WMMA GEMM -> fp32
// Workspace: Qb,Kb,Vb,Ob bf16 buffers = 4 * 16384*1024*2B = 128 MB.
// ---------------------------------------------------------------------------

typedef __attribute__((ext_vector_type(16))) __bf16 v16bf;
typedef __attribute__((ext_vector_type(8)))  float  v8f;
typedef int v4i_vs __attribute__((vector_size(16)));   // matches builtin proto

union FragB { uint4 u[2]; v16bf v; };

constexpr int H    = 16;
constexpr int DH   = 64;
constexpr int WIN  = 256;
constexpr int BTOK = 16384;   // B*N tokens
constexpr int DMODEL = 1024;  // query_dim == H*DH

#if defined(__has_builtin)
#if __has_builtin(__builtin_amdgcn_global_load_async_to_lds_b128) && \
    __has_builtin(__builtin_amdgcn_s_wait_asynccnt)
#define ATTN_ASYNC_LDS 1
#endif
#endif

// builtin signature: (AS1 v4i*, AS3 v4i*, imm offset, imm cpol)
#define TO_GLOBAL_V4(p) ((__attribute__((address_space(1))) v4i_vs*)(unsigned long long)(p))
#define TO_LDS_V4(p)    ((__attribute__((address_space(3))) v4i_vs*)(unsigned int)(unsigned long long)(p))

// ---------------------------------------------------------------------------
// Generic bf16 WMMA GEMM:  D[M,N] = A[M,K] * B[K,N] (+bias)
// A fp32 or bf16 (converted to bf16 tiles), B fp32 (converted), D fp32/bf16.
// Block tile 128x128, 256 threads = 8 waves (wave32), wave tile 32x64.
// Double-buffered LDS: fetch(kt+1) -> compute(kt) -> commit(kt+1) -> barrier.
// ---------------------------------------------------------------------------
template<bool A_BF16, bool OUT_BF16, bool BIAS>
__global__ __launch_bounds__(256)
void gemm_wmma(const void* __restrict__ Ap, const float* __restrict__ Bp,
               const float* __restrict__ bias, void* __restrict__ Dp,
               int M, int N, int K)
{
  __shared__ __align__(16) __bf16 As[2][128 * 32];   // [row][k]   2 x 8 KB
  __shared__ __align__(16) __bf16 Bs[2][128 * 32];   // [col][k]   2 x 8 KB

  const int tid  = threadIdx.x;
  const int lane = tid & 31;
  const int wid  = tid >> 5;
  const int wm   = wid & 3;    // 4 waves along M (32 rows each)
  const int wn   = wid >> 2;   // 2 waves along N (64 cols each)
  const int bn   = blockIdx.x;
  const int bm   = blockIdx.y;

  v8f acc[2][4] = {};

  const int nk = K >> 5;

  float4 fa[4];   // staged A tile (fp32 source)
  uint4  ba[2];   // staged A tile (bf16 source)
  float4 fb[4];   // staged B tile (fp32 source)

  auto fetch = [&](int kt) {
    if constexpr (A_BF16) {
      const __bf16* Ag = (const __bf16*)Ap;
      #pragma unroll
      for (int p = 0; p < 2; ++p) {
        int v = p * 256 + tid;
        int row = v >> 2, c8 = (v & 3) * 8;
        ba[p] = *(const uint4*)&Ag[(size_t)(bm * 128 + row) * K + kt * 32 + c8];
      }
    } else {
      const float* Ag = (const float*)Ap;
      #pragma unroll
      for (int p = 0; p < 4; ++p) {
        int v = p * 256 + tid;
        int row = v >> 3, c4 = (v & 7) * 4;
        fa[p] = *(const float4*)&Ag[(size_t)(bm * 128 + row) * K + kt * 32 + c4];
      }
    }
    #pragma unroll
    for (int p = 0; p < 4; ++p) {
      int v = p * 256 + tid;
      int kr = v >> 5, n4 = (v & 31) * 4;
      fb[p] = *(const float4*)&Bp[(size_t)(kt * 32 + kr) * N + bn * 128 + n4];
    }
  };

  auto commit = [&](int buf) {
    if constexpr (A_BF16) {
      #pragma unroll
      for (int p = 0; p < 2; ++p) {
        int v = p * 256 + tid;
        int row = v >> 2, c8 = (v & 3) * 8;
        *(uint4*)&As[buf][row * 32 + c8] = ba[p];
      }
    } else {
      #pragma unroll
      for (int p = 0; p < 4; ++p) {
        int v = p * 256 + tid;
        int row = v >> 3, c4 = (v & 7) * 4;
        __bf16* d = &As[buf][row * 32 + c4];
        d[0] = (__bf16)fa[p].x; d[1] = (__bf16)fa[p].y;
        d[2] = (__bf16)fa[p].z; d[3] = (__bf16)fa[p].w;
      }
    }
    #pragma unroll
    for (int p = 0; p < 4; ++p) {
      int v = p * 256 + tid;
      int kr = v >> 5, n4 = (v & 31) * 4;
      Bs[buf][(n4 + 0) * 32 + kr] = (__bf16)fb[p].x;
      Bs[buf][(n4 + 1) * 32 + kr] = (__bf16)fb[p].y;
      Bs[buf][(n4 + 2) * 32 + kr] = (__bf16)fb[p].z;
      Bs[buf][(n4 + 3) * 32 + kr] = (__bf16)fb[p].w;
    }
  };

  fetch(0);
  commit(0);
  __syncthreads();

  for (int kt = 0; kt < nk; ++kt) {
    const int cur = kt & 1;
    if (kt + 1 < nk) fetch(kt + 1);   // global loads overlap the WMMAs below

    #pragma unroll
    for (int mt = 0; mt < 2; ++mt) {
      FragB a;
      // A operand: elems 0..7 at k=8g, elems 8..15 at k=16+8g
      const __bf16* pa = &As[cur][(wm * 32 + mt * 16 + (lane & 15)) * 32 + 8 * (lane >> 4)];
      a.u[0] = *(const uint4*)pa;
      a.u[1] = *(const uint4*)(pa + 16);
      #pragma unroll
      for (int nt = 0; nt < 4; ++nt) {
        FragB b;
        // B operand: elems 0..7 at k=8g, elems 8..15 at k=16+8g for column n
        const __bf16* pb = &Bs[cur][(wn * 64 + nt * 16 + (lane & 15)) * 32 + 8 * (lane >> 4)];
        b.u[0] = *(const uint4*)pb;
        b.u[1] = *(const uint4*)(pb + 16);
        acc[mt][nt] = __builtin_amdgcn_wmma_f32_16x16x32_bf16(
            false, a.v, false, b.v, (short)0, acc[mt][nt], false, false);
      }
    }

    if (kt + 1 < nk) commit((kt + 1) & 1);
    __syncthreads();
  }

  // ---- writeback (C layout: vgpr r -> row r+8*(lane/16), col lane%16) ----
  #pragma unroll
  for (int mt = 0; mt < 2; ++mt)
    #pragma unroll
    for (int nt = 0; nt < 4; ++nt)
      #pragma unroll
      for (int r = 0; r < 8; ++r) {
        int ml = wm * 32 + mt * 16 + r + 8 * (lane >> 4);
        int nl = wn * 64 + nt * 16 + (lane & 15);
        size_t off = (size_t)(bm * 128 + ml) * N + bn * 128 + nl;
        float val = acc[mt][nt][r];
        if constexpr (BIAS) val += bias[bn * 128 + nl];
        if constexpr (OUT_BF16) ((__bf16*)Dp)[off] = (__bf16)val;
        else                    ((float*)Dp)[off]  = val;
      }
}

// ---------------------------------------------------------------------------
// Per-(window, head) attention: S = QK^T * 0.125, softmax rows, O = P V.
// One workgroup (8 waves) per block; each wave handles 2 passes of 16 rows.
// Un-normalized exp kept in a private fp32 LDS strip; 1/rowsum folded into
// the output write (avoids an in-place fp32->bf16 aliasing transpose).
// ---------------------------------------------------------------------------
__global__ __launch_bounds__(256)
void attn_win(const __bf16* __restrict__ Qb, const __bf16* __restrict__ Kb,
              const __bf16* __restrict__ Vb, __bf16* __restrict__ Ob)
{
  __shared__ __align__(16) __bf16 Qs[WIN * DH];     // 32 KB
  __shared__ __align__(16) __bf16 Ks[WIN * DH];     // 32 KB
  __shared__ __align__(16) __bf16 Vt[DH * WIN];     // 32 KB (transposed: [d][j])
  __shared__ __align__(16) float  Ss[8][16 * WIN];  // 128 KB (per-wave strip)
  __shared__ float invS[8][16];

  const int tid  = threadIdx.x;
  const int lane = tid & 31;
  const int wid  = tid >> 5;
  const int w    = blockIdx.x;   // window (64)
  const int h    = blockIdx.y;   // head (16)
  const float scale = 0.125f;    // DH^-0.5

  const uint4* Vg = (const uint4*)Vb;

#if ATTN_ASYNC_LDS
  // ---- Q,K: straight bf16 copies -> async DMA into LDS (ASYNCcnt) ----
  {
    const __bf16* qsrc = Qb + (size_t)(w * WIN) * DMODEL + h * DH;
    const __bf16* ksrc = Kb + (size_t)(w * WIN) * DMODEL + h * DH;
    #pragma unroll
    for (int p = 0; p < 8; ++p) {
      int v  = p * 256 + tid;
      int i  = v >> 3;          // token in window
      int dv = v & 7;           // 8-half group within head dim
      size_t goff = (size_t)i * DMODEL + dv * 8;
      __builtin_amdgcn_global_load_async_to_lds_b128(
          TO_GLOBAL_V4(qsrc + goff), TO_LDS_V4(&Qs[v * 8]), 0, 0);
      __builtin_amdgcn_global_load_async_to_lds_b128(
          TO_GLOBAL_V4(ksrc + goff), TO_LDS_V4(&Ks[v * 8]), 0, 0);
    }
  }
  // ---- V: needs transpose -> load via VGPRs, scatter to Vt ----
  #pragma unroll
  for (int p = 0; p < 8; ++p) {
    int v  = p * 256 + tid;
    int i  = v >> 3, dv = v & 7;
    uint4 x = Vg[(size_t)(w * WIN + i) * (DMODEL / 8) + h * 8 + dv];
    const __bf16* hx = (const __bf16*)&x;
    #pragma unroll
    for (int e = 0; e < 8; ++e) Vt[(dv * 8 + e) * WIN + i] = hx[e];
  }
  __builtin_amdgcn_s_wait_asynccnt(0);
  __syncthreads();
#else
  // ---- fallback: cooperative VGPR copies for Q, K, V ----
  {
    const uint4* Qg = (const uint4*)Qb;
    const uint4* Kg = (const uint4*)Kb;
    uint4* QsV = (uint4*)Qs;
    uint4* KsV = (uint4*)Ks;
    #pragma unroll
    for (int p = 0; p < 8; ++p) {
      int v  = p * 256 + tid;
      int i  = v >> 3, dv = v & 7;
      size_t g = (size_t)(w * WIN + i) * (DMODEL / 8) + h * 8 + dv;
      QsV[v] = Qg[g];
      KsV[v] = Kg[g];
      uint4 x = Vg[g];
      const __bf16* hx = (const __bf16*)&x;
      #pragma unroll
      for (int e = 0; e < 8; ++e) Vt[(dv * 8 + e) * WIN + i] = hx[e];
    }
  }
  __syncthreads();
#endif

  for (int pass = 0; pass < 2; ++pass) {
    const int rbase = wid * 32 + pass * 16;   // 16 query rows for this wave
    float* sw = Ss[wid];

    // ---- S strip: 16 x 256, WMMA over DH=64 (2 k-steps) ----
    #pragma unroll
    for (int nt = 0; nt < 16; ++nt) {
      v8f acc = {};
      #pragma unroll
      for (int kk = 0; kk < 64; kk += 32) {
        FragB a;
        const __bf16* pa = &Qs[(rbase + (lane & 15)) * DH + kk + 8 * (lane >> 4)];
        a.u[0] = *(const uint4*)pa;
        a.u[1] = *(const uint4*)(pa + 16);
        FragB b;   // B[k=d][n=j] = K[j][d] -> contiguous read of K row j
        const __bf16* pb = &Ks[(nt * 16 + (lane & 15)) * DH + kk + 8 * (lane >> 4)];
        b.u[0] = *(const uint4*)pb;
        b.u[1] = *(const uint4*)(pb + 16);
        acc = __builtin_amdgcn_wmma_f32_16x16x32_bf16(
            false, a.v, false, b.v, (short)0, acc, false, false);
      }
      #pragma unroll
      for (int r = 0; r < 8; ++r)
        sw[(r + 8 * (lane >> 4)) * WIN + nt * 16 + (lane & 15)] = acc[r] * scale;
    }

    // ---- softmax: 2 lanes per row (cols 0..127 / 128..255), shfl_xor(16) ----
    {
      int r  = lane & 15;
      int hf = lane >> 4;
      float* row = &sw[r * WIN + hf * 128];
      float mx = -3.0e38f;
      for (int c = 0; c < 128; c += 4) {
        float4 x = *(const float4*)&row[c];
        mx = fmaxf(mx, fmaxf(fmaxf(x.x, x.y), fmaxf(x.z, x.w)));
      }
      mx = fmaxf(mx, __shfl_xor(mx, 16, 32));
      float s = 0.f;
      for (int c = 0; c < 128; c += 4) {
        float4 x = *(float4*)&row[c];
        x.x = __expf(x.x - mx); x.y = __expf(x.y - mx);
        x.z = __expf(x.z - mx); x.w = __expf(x.w - mx);
        *(float4*)&row[c] = x;              // store un-normalized exp in place
        s += x.x + x.y + x.z + x.w;
      }
      s += __shfl_xor(s, 16, 32);
      if (hf == 0) invS[wid][r] = 1.0f / s;
    }
    __syncthreads();

    float inv[8];
    #pragma unroll
    for (int r = 0; r < 8; ++r) inv[r] = invS[wid][r + 8 * (lane >> 4)];

    // ---- O strip: 16 x 64 = E(16x256) * V(256x64), 8 k-steps of 32 ----
    #pragma unroll
    for (int nt = 0; nt < 4; ++nt) {
      v8f acc = {};
      #pragma unroll
      for (int ks = 0; ks < 8; ++ks) {
        // A operand from fp32 exp strip, converted at load time
        const float* pe = &sw[(lane & 15) * WIN + ks * 32 + 8 * (lane >> 4)];
        float4 x0 = *(const float4*)pe;
        float4 x1 = *(const float4*)(pe + 4);
        float4 x2 = *(const float4*)(pe + 16);
        float4 x3 = *(const float4*)(pe + 20);
        v16bf av;
        av[0]  = (__bf16)x0.x; av[1]  = (__bf16)x0.y;
        av[2]  = (__bf16)x0.z; av[3]  = (__bf16)x0.w;
        av[4]  = (__bf16)x1.x; av[5]  = (__bf16)x1.y;
        av[6]  = (__bf16)x1.z; av[7]  = (__bf16)x1.w;
        av[8]  = (__bf16)x2.x; av[9]  = (__bf16)x2.y;
        av[10] = (__bf16)x2.z; av[11] = (__bf16)x2.w;
        av[12] = (__bf16)x3.x; av[13] = (__bf16)x3.y;
        av[14] = (__bf16)x3.z; av[15] = (__bf16)x3.w;
        FragB b;   // B[k=j][n=d] = V[j][d] = Vt[d][j] -> contiguous read
        const __bf16* pb = &Vt[(nt * 16 + (lane & 15)) * WIN + ks * 32 + 8 * (lane >> 4)];
        b.u[0] = *(const uint4*)pb;
        b.u[1] = *(const uint4*)(pb + 16);
        acc = __builtin_amdgcn_wmma_f32_16x16x32_bf16(
            false, av, false, b.v, (short)0, acc, false, false);
      }
      #pragma unroll
      for (int r = 0; r < 8; ++r) {
        int ml = rbase + r + 8 * (lane >> 4);
        Ob[(size_t)(w * WIN + ml) * DMODEL + h * DH + nt * 16 + (lane & 15)] =
            (__bf16)(acc[r] * inv[r]);
      }
    }
    __syncthreads();
  }
}

// ---------------------------------------------------------------------------
extern "C" void kernel_launch(void* const* d_in, const int* in_sizes, int n_in,
                              void* d_out, int out_size, void* d_ws, size_t ws_size,
                              hipStream_t stream)
{
  const float* query = (const float*)d_in[0];
  // d_in[1] (context) is dead per the reference's overwrite bug.
  const float* Wq = (const float*)d_in[2];
  const float* Wk = (const float*)d_in[3];
  const float* Wv = (const float*)d_in[4];
  const float* Wo = (const float*)d_in[5];
  const float* bo = (const float*)d_in[6];

  const size_t tok = (size_t)BTOK * DMODEL;
  __bf16* Qb = (__bf16*)d_ws;        // 32 MB each
  __bf16* Kb = Qb + tok;
  __bf16* Vb = Kb + tok;
  __bf16* Ob = Vb + tok;

  dim3 gg(DMODEL / 128, BTOK / 128);   // (8, 128)
  dim3 tb(256);

  gemm_wmma<false, true, false><<<gg, tb, 0, stream>>>(query, Wq, nullptr, Qb, BTOK, DMODEL, DMODEL);
  gemm_wmma<false, true, false><<<gg, tb, 0, stream>>>(query, Wk, nullptr, Kb, BTOK, DMODEL, DMODEL);
  gemm_wmma<false, true, false><<<gg, tb, 0, stream>>>(query, Wv, nullptr, Vb, BTOK, DMODEL, DMODEL);

  attn_win<<<dim3(BTOK / WIN, H), tb, 0, stream>>>(Qb, Kb, Vb, Ob);

  gemm_wmma<true, false, true><<<gg, tb, 0, stream>>>(Ob, Wo, bo, d_out, BTOK, DMODEL, DMODEL);
}
